// DGCNN_35691178230210
// MI455X (gfx1250) — matrix operand order repaired and checked by reading, not compiled, not run on hardware
//
#include <hip/hip_runtime.h>
#include <hip/hip_fp16.h>

#define BB   16
#define NN   2048
#define KNNK 20
#define EPSF 1e-5f
#define SLOPE 0.2f

typedef __attribute__((ext_vector_type(16))) _Float16 v16h;
typedef __attribute__((ext_vector_type(8)))  float    v8f;

// ---------------------------------------------------------------------------
// transpose x [B,3,N] -> xt [B,N,3]
// ---------------------------------------------------------------------------
__global__ void k_transpose(const float* __restrict__ x, float* __restrict__ xt) {
    int i = blockIdx.x * blockDim.x + threadIdx.x;
    if (i >= BB * NN * 3) return;
    int c = i % 3;
    int n = (i / 3) % NN;
    int b = i / (3 * NN);
    xt[i] = x[((size_t)b * 3 + c) * NN + n];
}

// ---------------------------------------------------------------------------
// f32 [rows, C] (row stride srcStride) -> f16 [rows, Kpad], zero padded
// ---------------------------------------------------------------------------
__global__ void k_tof16(const float* __restrict__ src, int srcStride, int C, int Kpad,
                        _Float16* __restrict__ dst, int rows) {
    int i = blockIdx.x * blockDim.x + threadIdx.x;
    if (i >= rows * Kpad) return;
    int k = i % Kpad;
    int r = i / Kpad;
    float v = (k < C) ? src[(size_t)r * srcStride + k] : 0.0f;
    dst[i] = (_Float16)v;
}

// ---------------------------------------------------------------------------
// Edge-conv weight prep: W [O, 2C] f32 -> Wp [2O, Kpad] f16
//   rows 0..O-1   : W_a        = W[:, :C]
//   rows O..2O-1  : W_b - W_a  = W[:, C:] - W[:, :C]
// ---------------------------------------------------------------------------
__global__ void k_prepw(const float* __restrict__ W, int C, int O, int Kpad,
                        _Float16* __restrict__ Wp) {
    int i = blockIdx.x * blockDim.x + threadIdx.x;
    if (i >= 2 * O * Kpad) return;
    int k = i % Kpad;
    int r = i / Kpad;
    float v = 0.0f;
    if (k < C) {
        if (r < O) v = W[(size_t)r * 2 * C + k];
        else       v = W[(size_t)(r - O) * 2 * C + C + k] - W[(size_t)(r - O) * 2 * C + k];
    }
    Wp[i] = (_Float16)v;
}

// ---------------------------------------------------------------------------
// kNN (k=20, includes self): feat [B,N,C] strided, register top-k, LDS tiles
// ---------------------------------------------------------------------------
template <int C>
__global__ void k_knn(const float* __restrict__ feat, int ldf, int* __restrict__ idx) {
    const int QB = 128;   // queries per block
    const int TS = 32;    // candidate tile
    __shared__ float sm[TS * C];

    int b = blockIdx.x / (NN / QB);
    int q = (blockIdx.x % (NN / QB)) * QB + threadIdx.x;
    const float* fb = feat + (size_t)b * NN * ldf;

    float qf[C];
#pragma unroll
    for (int c = 0; c < C; c++) qf[c] = fb[(size_t)q * ldf + c];
    float sq = 0.0f;
#pragma unroll
    for (int c = 0; c < C; c++) sq += qf[c] * qf[c];

    float bd[KNNK];
    int   bi[KNNK];
#pragma unroll
    for (int j = 0; j < KNNK; j++) { bd[j] = -3.4e38f; bi[j] = 0; }
    float worst = -3.4e38f;
    int worsti = 0;

    for (int m0 = 0; m0 < NN; m0 += TS) {
        __syncthreads();
        for (int t = threadIdx.x; t < TS * C; t += QB)
            sm[t] = fb[(size_t)(m0 + t / C) * ldf + (t % C)];
        __syncthreads();
        for (int mm = 0; mm < TS; mm++) {
            const float* mf = &sm[mm * C];
            float dot = 0.0f, s2 = 0.0f;
#pragma unroll
            for (int c = 0; c < C; c++) { float v = mf[c]; dot += qf[c] * v; s2 += v * v; }
            float d = 2.0f * dot - sq - s2;   // neg squared distance
            if (d > worst) {
#pragma unroll
                for (int j = 0; j < KNNK; j++)
                    if (j == worsti) { bd[j] = d; bi[j] = m0 + mm; }
                worst = bd[0]; worsti = 0;
#pragma unroll
                for (int j = 1; j < KNNK; j++)
                    if (bd[j] < worst) { worst = bd[j]; worsti = j; }
            }
        }
    }
    int* op = idx + ((size_t)b * NN + q) * KNNK;
#pragma unroll
    for (int j = 0; j < KNNK; j++) op[j] = bi[j];
}

// ---------------------------------------------------------------------------
// WMMA fragment loads (wave32 layouts per CDNA5 ISA 7.12.2)
// A 16x32 f16: lanes 0-15 row M=lane, K {0..7,16..23}; lanes 16-31 K {8..15,24..31}
// B 32x16 f16: lane&15 = column N; lanes 0-15 K {0..7,16..23}; lanes 16-31 K {8..15,24..31}
// ---------------------------------------------------------------------------
__device__ __forceinline__ v16h load_fragA(const _Float16* __restrict__ A, int K,
                                           int row, int kk, int lane) {
    const _Float16* p = A + (size_t)row * K + kk + ((lane & 16) ? 8 : 0);
    v16h f;
#pragma unroll
    for (int i = 0; i < 8; i++) f[i] = p[i];
#pragma unroll
    for (int i = 0; i < 8; i++) f[8 + i] = p[16 + i];
    return f;
}

__device__ __forceinline__ v16h load_fragB(const _Float16* __restrict__ Wt, int K,
                                           int col, int kk, int lane) {
    const _Float16* p = Wt + (size_t)(col + (lane & 15)) * K + kk + ((lane & 16) ? 8 : 0);
    v16h f;
#pragma unroll
    for (int i = 0; i < 16; i++) f[i] = p[(i < 8) ? i : (i + 8)];
    return f;
}

#define WMMA_F16(acc, a, b) \
    acc = __builtin_amdgcn_wmma_f32_16x16x32_f16(false, a, false, b, (short)0, acc, false, false)

// ---------------------------------------------------------------------------
// GEMM: D[M,Nout] f32 = A[M,K] f16 x Wt[Nout,K]^T f16.
// One wave -> 32x64 tile (2 A fragments share 4 B fragments: 8 WMMA / K-step).
// ---------------------------------------------------------------------------
__global__ void k_gemm(const _Float16* __restrict__ A, const _Float16* __restrict__ Wt,
                       float* __restrict__ D, int M, int Nout, int K) {
    int lane = threadIdx.x;
    int m0 = blockIdx.x * 32;
    int n0 = blockIdx.y * 64;
    v8f p00 = {}, p01 = {}, p02 = {}, p03 = {};
    v8f p10 = {}, p11 = {}, p12 = {}, p13 = {};
    int rowA0 = m0 + (lane & 15);
    int rowA1 = rowA0 + 16;
    const _Float16* pf0 = A + (size_t)rowA0 * K;
    const _Float16* pf1 = A + (size_t)rowA1 * K;
    for (int kk = 0; kk < K; kk += 32) {
        __builtin_prefetch(pf0 + kk + 64, 0, 3);   // near-scope speculative
        __builtin_prefetch(pf1 + kk + 64, 0, 3);
        v16h a0 = load_fragA(A, K, rowA0, kk, lane);
        v16h a1 = load_fragA(A, K, rowA1, kk, lane);
        v16h b0 = load_fragB(Wt, K, n0,      kk, lane);
        v16h b1 = load_fragB(Wt, K, n0 + 16, kk, lane);
        v16h b2 = load_fragB(Wt, K, n0 + 32, kk, lane);
        v16h b3 = load_fragB(Wt, K, n0 + 48, kk, lane);
        WMMA_F16(p00, a0, b0); WMMA_F16(p10, a1, b0);
        WMMA_F16(p01, a0, b1); WMMA_F16(p11, a1, b1);
        WMMA_F16(p02, a0, b2); WMMA_F16(p12, a1, b2);
        WMMA_F16(p03, a0, b3); WMMA_F16(p13, a1, b3);
    }
    int rr = (lane & 16) ? 8 : 0;
    int cc = lane & 15;
#pragma unroll
    for (int r = 0; r < 8; r++) {
        float* dp0 = D + (size_t)(m0 + rr + r) * Nout + n0 + cc;
        dp0[0]  = p00[r];
        dp0[16] = p01[r];
        dp0[32] = p02[r];
        dp0[48] = p03[r];
        float* dp1 = D + (size_t)(m0 + 16 + rr + r) * Nout + n0 + cc;
        dp1[0]  = p10[r];
        dp1[16] = p11[r];
        dp1[32] = p12[r];
        dp1[48] = p13[r];
    }
}

// ---------------------------------------------------------------------------
// Layer-5 GEMM fused with BN + LeakyReLU + max/sum pooling over N.
// grid.x = B*16 rowblocks (128 rows each), grid.y = 1024/64, block = 32.
// Inner loop processes 32 rows per pass so weight fragments feed 2 WMMAs each.
// ---------------------------------------------------------------------------
__global__ void k_gemm_pool(const _Float16* __restrict__ A, const _Float16* __restrict__ W5,
                            const float* __restrict__ bn5,
                            float* __restrict__ pmax, float* __restrict__ psum) {
    const int K = 512;
    int lane = threadIdx.x;
    int b  = blockIdx.x / 16;
    int rb = blockIdx.x % 16;
    int n0 = blockIdx.y * 64;
    int cc = lane & 15;

    float s[4], be[4], mu[4];
#pragma unroll
    for (int t = 0; t < 4; t++) {
        int o = n0 + t * 16 + cc;
        s[t]  = bn5[o] * rsqrtf(bn5[3072 + o] + EPSF);
        be[t] = bn5[1024 + o];
        mu[t] = bn5[2048 + o];
    }

    float cmax[4] = {-3.4e38f, -3.4e38f, -3.4e38f, -3.4e38f};
    float csum[4] = {0.f, 0.f, 0.f, 0.f};

    for (int rt = 0; rt < 4; rt++) {
        int m0 = b * NN + rb * 128 + rt * 32;
        v8f p00 = {}, p01 = {}, p02 = {}, p03 = {};
        v8f p10 = {}, p11 = {}, p12 = {}, p13 = {};
        int rowA0 = m0 + (lane & 15);
        int rowA1 = rowA0 + 16;
        const _Float16* pf0 = A + (size_t)rowA0 * K;
        const _Float16* pf1 = A + (size_t)rowA1 * K;
        for (int kk = 0; kk < K; kk += 32) {
            __builtin_prefetch(pf0 + kk + 64, 0, 3);
            __builtin_prefetch(pf1 + kk + 64, 0, 3);
            v16h a0 = load_fragA(A, K, rowA0, kk, lane);
            v16h a1 = load_fragA(A, K, rowA1, kk, lane);
            v16h b0 = load_fragB(W5, K, n0,      kk, lane);
            v16h b1 = load_fragB(W5, K, n0 + 16, kk, lane);
            v16h b2 = load_fragB(W5, K, n0 + 32, kk, lane);
            v16h b3 = load_fragB(W5, K, n0 + 48, kk, lane);
            WMMA_F16(p00, a0, b0); WMMA_F16(p10, a1, b0);
            WMMA_F16(p01, a0, b1); WMMA_F16(p11, a1, b1);
            WMMA_F16(p02, a0, b2); WMMA_F16(p12, a1, b2);
            WMMA_F16(p03, a0, b3); WMMA_F16(p13, a1, b3);
        }
#pragma unroll
        for (int r = 0; r < 8; r++) {
            float h;
            h = (p00[r] - mu[0]) * s[0] + be[0]; h = h >= 0.f ? h : SLOPE * h;
            cmax[0] = fmaxf(cmax[0], h); csum[0] += h;
            h = (p10[r] - mu[0]) * s[0] + be[0]; h = h >= 0.f ? h : SLOPE * h;
            cmax[0] = fmaxf(cmax[0], h); csum[0] += h;
            h = (p01[r] - mu[1]) * s[1] + be[1]; h = h >= 0.f ? h : SLOPE * h;
            cmax[1] = fmaxf(cmax[1], h); csum[1] += h;
            h = (p11[r] - mu[1]) * s[1] + be[1]; h = h >= 0.f ? h : SLOPE * h;
            cmax[1] = fmaxf(cmax[1], h); csum[1] += h;
            h = (p02[r] - mu[2]) * s[2] + be[2]; h = h >= 0.f ? h : SLOPE * h;
            cmax[2] = fmaxf(cmax[2], h); csum[2] += h;
            h = (p12[r] - mu[2]) * s[2] + be[2]; h = h >= 0.f ? h : SLOPE * h;
            cmax[2] = fmaxf(cmax[2], h); csum[2] += h;
            h = (p03[r] - mu[3]) * s[3] + be[3]; h = h >= 0.f ? h : SLOPE * h;
            cmax[3] = fmaxf(cmax[3], h); csum[3] += h;
            h = (p13[r] - mu[3]) * s[3] + be[3]; h = h >= 0.f ? h : SLOPE * h;
            cmax[3] = fmaxf(cmax[3], h); csum[3] += h;
        }
    }
    // combine row halves: lanes L and L+16 hold the same columns
#pragma unroll
    for (int t = 0; t < 4; t++) {
        float om = __shfl_xor(cmax[t], 16, 32);
        float os = __shfl_xor(csum[t], 16, 32);
        cmax[t] = fmaxf(cmax[t], om);
        csum[t] += os;
    }
    if (lane < 16) {
#pragma unroll
        for (int t = 0; t < 4; t++) {
            int o = n0 + t * 16 + cc;
            pmax[((size_t)b * 16 + rb) * 1024 + o] = cmax[t];
            psum[((size_t)b * 16 + rb) * 1024 + o] = csum[t];
        }
    }
}

__global__ void k_pool_reduce(const float* __restrict__ pmax, const float* __restrict__ psum,
                              float* __restrict__ pooled) {
    int i = blockIdx.x * blockDim.x + threadIdx.x;
    if (i >= BB * 2048) return;
    int b = i / 2048;
    int c = i % 2048;
    if (c < 1024) {
        float m = -3.4e38f;
        for (int rb = 0; rb < 16; rb++) m = fmaxf(m, pmax[((size_t)b * 16 + rb) * 1024 + c]);
        pooled[i] = m;
    } else {
        int o = c - 1024;
        float ssum = 0.f;
        for (int rb = 0; rb < 16; rb++) ssum += psum[((size_t)b * 16 + rb) * 1024 + o];
        pooled[i] = ssum / (float)NN;
    }
}

// ---------------------------------------------------------------------------
// Edge combine: out[b,n,o] = max_k leaky(bn(Y[b,idx[k],o] + Z[b,n,o]))
// yz layout [B*N, 2O]: cols [0,O) = Y (neighbor term), [O,2O) = Z (center term)
// ---------------------------------------------------------------------------
__global__ void k_edge_combine(const float* __restrict__ yz, const int* __restrict__ idx,
                               const float* __restrict__ bnp, int O,
                               float* __restrict__ out, int outStride, int colBase) {
    int bn_ = blockIdx.x;          // b*N + n
    int b = bn_ / NN;
    int o = threadIdx.x;
    __shared__ int sidx[KNNK];
    if (threadIdx.x < KNNK) sidx[threadIdx.x] = idx[(size_t)bn_ * KNNK + threadIdx.x];
    __syncthreads();

    int twoO = 2 * O;
    float z  = yz[(size_t)bn_ * twoO + O + o];
    float sc = bnp[o] * rsqrtf(bnp[3 * O + o] + EPSF);
    float bb = bnp[O + o];
    float mu = bnp[2 * O + o];
    float mx = -3.4e38f;
#pragma unroll 4
    for (int k = 0; k < KNNK; k++) {
        float y = yz[((size_t)b * NN + sidx[k]) * twoO + o];
        float h = (y + z - mu) * sc + bb;
        h = h >= 0.f ? h : SLOPE * h;
        mx = fmaxf(mx, h);
    }
    out[(size_t)bn_ * outStride + colBase + o] = mx;
}

// ---------------------------------------------------------------------------
// Small FC: out[b,o] = act(bn((in[b,:] . W[o,:]) + bias))
// ---------------------------------------------------------------------------
__global__ void k_fc(const float* __restrict__ in, const float* __restrict__ W,
                     const float* __restrict__ bias, const float* __restrict__ bnp,
                     int I, int O, float* __restrict__ out) {
    int i = blockIdx.x * blockDim.x + threadIdx.x;
    if (i >= BB * O) return;
    int b = i / O;
    int o = i % O;
    const float* ip = in + (size_t)b * I;
    const float* wp = W + (size_t)o * I;
    float acc = 0.f;
    for (int k = 0; k < I; k++) acc += ip[k] * wp[k];
    if (bias) acc += bias[o];
    if (bnp) {
        acc = (acc - bnp[2 * O + o]) * (bnp[o] * rsqrtf(bnp[3 * O + o] + EPSF)) + bnp[O + o];
        acc = acc >= 0.f ? acc : SLOPE * acc;
    }
    out[i] = acc;
}

// ---------------------------------------------------------------------------
// Orchestration
// ---------------------------------------------------------------------------
extern "C" void kernel_launch(void* const* d_in, const int* in_sizes, int n_in,
                              void* d_out, int out_size, void* d_ws, size_t ws_size,
                              hipStream_t stream) {
    (void)in_sizes; (void)n_in; (void)out_size; (void)ws_size;
    const float* x   = (const float*)d_in[0];
    const float* W1  = (const float*)d_in[1];
    const float* bn1 = (const float*)d_in[2];
    const float* W2  = (const float*)d_in[3];
    const float* bn2 = (const float*)d_in[4];
    const float* W3  = (const float*)d_in[5];
    const float* bn3 = (const float*)d_in[6];
    const float* W4  = (const float*)d_in[7];
    const float* bn4 = (const float*)d_in[8];
    const float* W5  = (const float*)d_in[9];
    const float* bn5 = (const float*)d_in[10];
    const float* Wl1 = (const float*)d_in[11];
    const float* bn6 = (const float*)d_in[12];
    const float* Wl2 = (const float*)d_in[13];
    const float* bl2 = (const float*)d_in[14];
    const float* bn7 = (const float*)d_in[15];
    const float* Wl3 = (const float*)d_in[16];
    const float* bl3 = (const float*)d_in[17];

    char* ws = (char*)d_ws;
    constexpr size_t SZ_XT  = (size_t)BB * NN * 3 * 4;          //   0.4 MB
    constexpr size_t SZ_CAT = (size_t)BB * NN * 512 * 4;        //  67 MB
    constexpr size_t SZ_FA  = (size_t)BB * NN * 512 * 2;        //  33.5 MB
    constexpr size_t SZ_YZ  = (size_t)BB * NN * 512 * 4;        //  67 MB
    constexpr size_t SZ_IDX = (size_t)BB * NN * KNNK * 4;       //   2.6 MB
    constexpr size_t SZ_WE  = (size_t)2 * 256 * 128 * 2;        //   128 KB
    constexpr size_t SZ_W5  = (size_t)1024 * 512 * 2;           //     1 MB
    constexpr size_t SZ_PM  = (size_t)BB * 16 * 1024 * 4;       //     1 MB
    constexpr size_t SZ_PL  = (size_t)BB * 2048 * 4;
    constexpr size_t SZ_H1  = (size_t)BB * 512 * 4;

    size_t off = 0;
    float*    xt   = (float*)(ws + off);    off += SZ_XT;
    float*    cat  = (float*)(ws + off);    off += SZ_CAT;
    _Float16* fa   = (_Float16*)(ws + off); off += SZ_FA;
    float*    yz   = (float*)(ws + off);    off += SZ_YZ;
    int*      idxb = (int*)(ws + off);      off += SZ_IDX;
    _Float16* we   = (_Float16*)(ws + off); off += SZ_WE;
    _Float16* w5h  = (_Float16*)(ws + off); off += SZ_W5;
    float*    pmax = (float*)(ws + off);    off += SZ_PM;
    float*    psum = (float*)(ws + off);    off += SZ_PM;
    float*    pool = (float*)(ws + off);    off += SZ_PL;
    float*    h1   = (float*)(ws + off);    off += SZ_H1;
    float*    h2   = (float*)(ws + off);

    const int M = BB * NN;                 // 32768
    const int knnBlocks = BB * (NN / 128); // 256

    // xt
    k_transpose<<<(BB * NN * 3 + 255) / 256, 256, 0, stream>>>(x, xt);

    // ---- EdgeConv 1: C=3 (Kpad 32), O=64, out -> cat cols [0,64)
    k_knn<3><<<knnBlocks, 128, 0, stream>>>(xt, 3, idxb);
    k_tof16<<<(M * 32 + 255) / 256, 256, 0, stream>>>(xt, 3, 3, 32, fa, M);
    k_prepw<<<(2 * 64 * 32 + 255) / 256, 256, 0, stream>>>(W1, 3, 64, 32, we);
    k_gemm<<<dim3(M / 32, 128 / 64), 32, 0, stream>>>(fa, we, yz, M, 128, 32);
    k_edge_combine<<<M, 64, 0, stream>>>(yz, idxb, bn1, 64, cat, 512, 0);

    // ---- EdgeConv 2: C=64, O=64, in cat[:,0:64), out -> cat cols [64,128)
    k_knn<64><<<knnBlocks, 128, 0, stream>>>(cat, 512, idxb);
    k_tof16<<<(M * 64 + 255) / 256, 256, 0, stream>>>(cat, 512, 64, 64, fa, M);
    k_prepw<<<(2 * 64 * 64 + 255) / 256, 256, 0, stream>>>(W2, 64, 64, 64, we);
    k_gemm<<<dim3(M / 32, 128 / 64), 32, 0, stream>>>(fa, we, yz, M, 128, 64);
    k_edge_combine<<<M, 64, 0, stream>>>(yz, idxb, bn2, 64, cat, 512, 64);

    // ---- EdgeConv 3: C=64, O=128, in cat[:,64:128), out -> cat cols [128,256)
    k_knn<64><<<knnBlocks, 128, 0, stream>>>(cat + 64, 512, idxb);
    k_tof16<<<(M * 64 + 255) / 256, 256, 0, stream>>>(cat + 64, 512, 64, 64, fa, M);
    k_prepw<<<(2 * 128 * 64 + 255) / 256, 256, 0, stream>>>(W3, 64, 128, 64, we);
    k_gemm<<<dim3(M / 32, 256 / 64), 32, 0, stream>>>(fa, we, yz, M, 256, 64);
    k_edge_combine<<<M, 128, 0, stream>>>(yz, idxb, bn3, 128, cat, 512, 128);

    // ---- EdgeConv 4: C=128, O=256, in cat[:,128:256), out -> cat cols [256,512)
    k_knn<128><<<knnBlocks, 128, 0, stream>>>(cat + 128, 512, idxb);
    k_tof16<<<(M * 128 + 255) / 256, 256, 0, stream>>>(cat + 128, 512, 128, 128, fa, M);
    k_prepw<<<(2 * 256 * 128 + 255) / 256, 256, 0, stream>>>(W4, 128, 256, 128, we);
    k_gemm<<<dim3(M / 32, 512 / 64), 32, 0, stream>>>(fa, we, yz, M, 512, 128);
    k_edge_combine<<<M, 256, 0, stream>>>(yz, idxb, bn4, 256, cat, 512, 256);

    // ---- Layer 5: cat [B*N,512] x W5[1024,512]^T, fused BN/act + max/avg pool
    k_tof16<<<(M * 512 + 255) / 256, 256, 0, stream>>>(cat, 512, 512, 512, fa, M);
    k_tof16<<<(1024 * 512 + 255) / 256, 256, 0, stream>>>(W5, 512, 512, 512, w5h, 1024);
    k_gemm_pool<<<dim3(BB * 16, 1024 / 64), 32, 0, stream>>>(fa, w5h, bn5, pmax, psum);
    k_pool_reduce<<<(BB * 2048 + 255) / 256, 256, 0, stream>>>(pmax, psum, pool);

    // ---- FC head
    k_fc<<<(BB * 512 + 255) / 256, 256, 0, stream>>>(pool, Wl1, nullptr, bn6, 2048, 512, h1);
    k_fc<<<(BB * 256 + 255) / 256, 256, 0, stream>>>(h1, Wl2, bl2, bn7, 512, 256, h2);
    k_fc<<<(BB * 40 + 255) / 256, 256, 0, stream>>>(h2, Wl3, bl3, nullptr, 256, 40, (float*)d_out);
}
